// MSpatialGCN_13194139533745
// MI455X (gfx1250) — matrix operand order repaired and checked by reading, not compiled, not run on hardware
//
#include <hip/hip_runtime.h>

#define NTOT 3136            // 56*56
#define EPS  1e-5f

typedef __attribute__((ext_vector_type(16))) __bf16 v16bf;
typedef __attribute__((ext_vector_type(8)))  float  v8f;

__device__ __forceinline__ unsigned short f2bf(float f) {
    unsigned u = __float_as_uint(f);
    u += 0x7FFFu + ((u >> 16) & 1u);     // round-to-nearest-even
    return (unsigned short)(u >> 16);
}

// A operand, 16x32 bf16 (M x K), from row-major [row][K] array (rowStride elems).
// ISA layout: lanes 0-15: M=lane, K 0-7 (v0-3) + 16-23 (v4-7); lanes 16-31: M=lane-16, K 8-15 + 24-31.
__device__ __forceinline__ v16bf load_a16(const unsigned short* base, int lane, int rowStride) {
    int row = lane & 15;
    int kb  = (lane >> 4) * 8;
    const unsigned short* p = base + (size_t)row * rowStride + kb;
    union { uint4 u[2]; v16bf v; } t;
    t.u[0] = *(const uint4*)(p);
    t.u[1] = *(const uint4*)(p + 16);
    return t.v;
}

// B operand, 32x16 bf16 (K x N), from column-major [col][K] array (colStride elems).
// ISA layout: lanes 0-15: col=lane, K 0-7 + 16-23; lanes 16-31: col=lane-16, K 8-15 + 24-31.
__device__ __forceinline__ v16bf load_b16(const unsigned short* base, int lane, int colStride) {
    int col = lane & 15;
    int kb  = (lane >> 4) * 8;
    const unsigned short* p = base + (size_t)col * colStride + kb;
    union { uint4 u[2]; v16bf v; } t;
    t.u[0] = *(const uint4*)(p);
    t.u[1] = *(const uint4*)(p + 16);
    return t.v;
}

__device__ __forceinline__ v8f wmma_bf16(v16bf a, v16bf b, v8f c) {
    return __builtin_amdgcn_wmma_f32_16x16x32_bf16(false, a, false, b, (short)0, c, false, false);
}

// ---------------- Kernel 1: QKV 1x1 conv, emit bf16 K, Q^T ([bh][n][32]) and f32 V ([b][128][N]) --
__global__ void k_qkv(const float* __restrict__ x,
                      const float* __restrict__ wk, const float* __restrict__ bk,
                      const float* __restrict__ wq, const float* __restrict__ bq,
                      const float* __restrict__ wv, const float* __restrict__ bv,
                      unsigned short* __restrict__ kb, unsigned short* __restrict__ qb,
                      float* __restrict__ vf) {
    int n   = blockIdx.x * 256 + threadIdx.x;
    int bch = blockIdx.y;                 // b*128 + ch
    if (n >= NTOT) return;
    int b = bch >> 7, ch = bch & 127;
    const float* xp = x + (size_t)(b * 64) * NTOT + n;
    float ak = bk[ch], aq = bq[ch], av = bv[ch];
    #pragma unroll 8
    for (int c = 0; c < 64; ++c) {
        float xv = xp[(size_t)c * NTOT];
        ak += wk[ch * 64 + c] * xv;
        aq += wq[ch * 64 + c] * xv;
        av += wv[ch * 64 + c] * xv;
    }
    int h = ch >> 5, d = ch & 31;
    size_t o = ((size_t)(b * 4 + h) * NTOT + n) * 32 + d;
    kb[o] = f2bf(ak);
    qb[o] = f2bf(aq);
    vf[(size_t)bch * NTOT + n] = av;
}

// ---------------- Kernel 2: column-wise softmax stats (max/sum over n, per column m) -------------
__global__ void k_colstats(const unsigned short* __restrict__ kb,
                           const unsigned short* __restrict__ qb,
                           float* __restrict__ colmax, float* __restrict__ colsum) {
    int lane = threadIdx.x & 31, wv = threadIdx.x >> 5;
    int bh = blockIdx.y;
    int m0 = (blockIdx.x * 4 + wv) * 16;
    const unsigned short* qbase = qb + ((size_t)bh * NTOT + m0) * 32;
    const unsigned short* kbase = kb + (size_t)bh * NTOT * 32;
    v16bf bop = load_b16(qbase, lane, 32);            // reused across all n tiles
    float mx = -3.0e38f, sm = 0.f;
    for (int nt = 0; nt < NTOT / 16; ++nt) {
        v16bf aop = load_a16(kbase + (size_t)nt * 16 * 32, lane, 32);
        v8f c = {};
        c = wmma_bf16(aop, bop, c);
        float tm = c[0];
        #pragma unroll
        for (int r = 1; r < 8; ++r) tm = fmaxf(tm, c[r]);
        float nm = fmaxf(mx, tm);
        float s = 0.f;
        #pragma unroll
        for (int r = 0; r < 8; ++r) s += __expf(c[r] - nm);
        sm = sm * __expf(mx - nm) + s;
        mx = nm;
    }
    float mx2 = __shfl_xor(mx, 16, 32);               // merge the two row-halves of the column
    float sm2 = __shfl_xor(sm, 16, 32);
    float M = fmaxf(mx, mx2);
    float S = sm * __expf(mx - M) + sm2 * __expf(mx2 - M);
    if (lane < 16) {
        colmax[(size_t)bh * NTOT + m0 + lane] = M;
        colsum[(size_t)bh * NTOT + m0 + lane] = S;
    }
}

// ---------------- Kernel 3: V' = bf16(V / colsum), channel-major [b][128][N] ---------------------
__global__ void k_vscale(const float* __restrict__ vf, const float* __restrict__ colsum,
                         unsigned short* __restrict__ vtb) {
    int n = blockIdx.x * 256 + threadIdx.x;
    int bch = blockIdx.y;
    if (n >= NTOT) return;
    int b = bch >> 7, h = (bch & 127) >> 5;
    float s = colsum[(size_t)(b * 4 + h) * NTOT + n];
    vtb[(size_t)bch * NTOT + n] = f2bf(vf[(size_t)bch * NTOT + n] / s);
}

// ---------------- Kernel 4: av = softmax(adj) @ V, fused (recompute adj tiles) -------------------
// out: av_t f32 [bh][32][N]
__global__ void k_av(const unsigned short* __restrict__ kb, const unsigned short* __restrict__ qb,
                     const unsigned short* __restrict__ vtb, const float* __restrict__ colmax,
                     float* __restrict__ av_t) {
    __shared__ unsigned short smP[4][16 * 32];
    int lane = threadIdx.x & 31, wv = threadIdx.x >> 5;
    int bh = blockIdx.y, b = bh >> 2, h = bh & 3;
    int n0 = (blockIdx.x * 4 + wv) * 16;
    const unsigned short* kbase = kb + ((size_t)bh * NTOT + n0) * 32;
    const unsigned short* qbase = qb + (size_t)bh * NTOT * 32;
    const unsigned short* vbase = vtb + (size_t)(b * 128 + h * 32) * NTOT;
    const float* cmax = colmax + (size_t)bh * NTOT;
    unsigned short* sp = smP[wv];
    v16bf ak = load_a16(kbase, lane, 32);             // K rows fixed for this wave
    v8f c0 = {}, c1 = {};
    int col = lane & 15, rbase = (lane >> 4) * 8;
    for (int mt = 0; mt < NTOT / 32; ++mt) {
        int m0 = mt * 32;
        v16bf bq1 = load_b16(qbase + (size_t)m0 * 32, lane, 32);
        v16bf bq2 = load_b16(qbase + (size_t)(m0 + 16) * 32, lane, 32);
        v8f t1 = {}, t2 = {};
        t1 = wmma_bf16(ak, bq1, t1);
        t2 = wmma_bf16(ak, bq2, t2);
        float cm1 = cmax[m0 + col], cm2 = cmax[m0 + 16 + col];
        #pragma unroll
        for (int r = 0; r < 8; ++r) {                 // C-layout -> row-major bf16 P tile in LDS
            sp[(rbase + r) * 32 + col]      = f2bf(__expf(t1[r] - cm1));
            sp[(rbase + r) * 32 + 16 + col] = f2bf(__expf(t2[r] - cm2));
        }
        asm volatile("s_wait_dscnt 0" ::: "memory");  // same-wave LDS RAW fence
        v16bf ap  = load_a16(sp, lane, 32);           // reload as A operand (16n x 32m)
        v16bf bv0 = load_b16(vbase + m0, lane, NTOT);
        v16bf bv1 = load_b16(vbase + (size_t)16 * NTOT + m0, lane, NTOT);
        c0 = wmma_bf16(ap, bv0, c0);
        c1 = wmma_bf16(ap, bv1, c1);
    }
    float* outb = av_t + (size_t)bh * 32 * NTOT;
    int nlo = n0 + rbase;
    *(float4*)(outb + (size_t)col * NTOT + nlo)            = make_float4(c0[0], c0[1], c0[2], c0[3]);
    *(float4*)(outb + (size_t)col * NTOT + nlo + 4)        = make_float4(c0[4], c0[5], c0[6], c0[7]);
    *(float4*)(outb + (size_t)(col + 16) * NTOT + nlo)     = make_float4(c1[0], c1[1], c1[2], c1[3]);
    *(float4*)(outb + (size_t)(col + 16) * NTOT + nlo + 4) = make_float4(c1[4], c1[5], c1[6], c1[7]);
}

// ---------------- Kernel 5: avw = relu(bn1(av @ w1^T + b1)) / colsum -> bf16 [b][256ch][N] -------
__global__ void k_avw(const float* __restrict__ av_t, const float* __restrict__ w1,
                      const float* __restrict__ b1, const float* __restrict__ g1,
                      const float* __restrict__ be1, const float* __restrict__ rm1,
                      const float* __restrict__ rv1, const float* __restrict__ colsum,
                      unsigned short* __restrict__ awt) {
    int m = blockIdx.x * 256 + threadIdx.x;
    int idx = blockIdx.y;                 // b*256 + h*64 + o
    if (m >= NTOT) return;
    int b = idx >> 8, chh = idx & 255, h = chh >> 6, o = chh & 63;
    const float* ab = av_t + (size_t)(b * 4 + h) * 32 * NTOT + m;
    float acc = b1[o];
    #pragma unroll 8
    for (int d = 0; d < 32; ++d) acc += ab[(size_t)d * NTOT] * w1[o * 32 + d];
    float sc  = g1[chh] * rsqrtf(rv1[chh] + EPS);
    float val = fmaxf((acc - rm1[chh]) * sc + be1[chh], 0.f);
    float cs  = colsum[(size_t)(b * 4 + h) * NTOT + m];
    awt[(size_t)idx * NTOT + m] = f2bf(val / cs);
}

// ---------------- Kernel 6: av2 = softmax(adj) @ avw, fused -> f32 [bh][64][N] -------------------
__global__ void k_av2(const unsigned short* __restrict__ kb, const unsigned short* __restrict__ qb,
                      const unsigned short* __restrict__ awt, const float* __restrict__ colmax,
                      float* __restrict__ av2_t) {
    __shared__ unsigned short smP[4][16 * 32];
    int lane = threadIdx.x & 31, wv = threadIdx.x >> 5;
    int bh = blockIdx.y, b = bh >> 2, h = bh & 3;
    int n0 = (blockIdx.x * 4 + wv) * 16;
    const unsigned short* kbase = kb + ((size_t)bh * NTOT + n0) * 32;
    const unsigned short* qbase = qb + (size_t)bh * NTOT * 32;
    const unsigned short* wbase = awt + (size_t)(b * 256 + h * 64) * NTOT;
    const float* cmax = colmax + (size_t)bh * NTOT;
    unsigned short* sp = smP[wv];
    v16bf ak = load_a16(kbase, lane, 32);
    v8f c0 = {}, c1 = {}, c2 = {}, c3 = {};
    int col = lane & 15, rbase = (lane >> 4) * 8;
    for (int mt = 0; mt < NTOT / 32; ++mt) {
        int m0 = mt * 32;
        v16bf bq1 = load_b16(qbase + (size_t)m0 * 32, lane, 32);
        v16bf bq2 = load_b16(qbase + (size_t)(m0 + 16) * 32, lane, 32);
        v8f t1 = {}, t2 = {};
        t1 = wmma_bf16(ak, bq1, t1);
        t2 = wmma_bf16(ak, bq2, t2);
        float cm1 = cmax[m0 + col], cm2 = cmax[m0 + 16 + col];
        #pragma unroll
        for (int r = 0; r < 8; ++r) {
            sp[(rbase + r) * 32 + col]      = f2bf(__expf(t1[r] - cm1));
            sp[(rbase + r) * 32 + 16 + col] = f2bf(__expf(t2[r] - cm2));
        }
        asm volatile("s_wait_dscnt 0" ::: "memory");
        v16bf ap = load_a16(sp, lane, 32);
        c0 = wmma_bf16(ap, load_b16(wbase + m0, lane, NTOT), c0);
        c1 = wmma_bf16(ap, load_b16(wbase + (size_t)16 * NTOT + m0, lane, NTOT), c1);
        c2 = wmma_bf16(ap, load_b16(wbase + (size_t)32 * NTOT + m0, lane, NTOT), c2);
        c3 = wmma_bf16(ap, load_b16(wbase + (size_t)48 * NTOT + m0, lane, NTOT), c3);
    }
    float* outb = av2_t + (size_t)bh * 64 * NTOT;
    int nlo = n0 + rbase;
    v8f cc[4] = {c0, c1, c2, c3};
    #pragma unroll
    for (int q = 0; q < 4; ++q) {
        float* op = outb + (size_t)(col + 16 * q) * NTOT + nlo;
        *(float4*)(op)     = make_float4(cc[q][0], cc[q][1], cc[q][2], cc[q][3]);
        *(float4*)(op + 4) = make_float4(cc[q][4], cc[q][5], cc[q][6], cc[q][7]);
    }
}

// ---------------- Kernel 7: y = relu(bn2(av2 @ w2^T + b2)) -> f32 [b][256ch][N] ------------------
__global__ void k_avw2(const float* __restrict__ av2_t, const float* __restrict__ w2,
                       const float* __restrict__ b2, const float* __restrict__ g2,
                       const float* __restrict__ be2, const float* __restrict__ rm2,
                       const float* __restrict__ rv2, float* __restrict__ y) {
    int m = blockIdx.x * 256 + threadIdx.x;
    int idx = blockIdx.y;                 // b*256 + h*64 + o
    if (m >= NTOT) return;
    int b = idx >> 8, chh = idx & 255, h = chh >> 6, o = chh & 63;
    const float* ab = av2_t + (size_t)(b * 4 + h) * 64 * NTOT + m;
    float acc = b2[o];
    #pragma unroll 8
    for (int d = 0; d < 64; ++d) acc += ab[(size_t)d * NTOT] * w2[o * 64 + d];
    float sc = g2[chh] * rsqrtf(rv2[chh] + EPS);
    y[(size_t)idx * NTOT + m] = fmaxf((acc - rm2[chh]) * sc + be2[chh], 0.f);
}

// ---------------- Kernel 8: final 1x1 projection out = wf @ y + bf -------------------------------
__global__ void k_final(const float* __restrict__ y, const float* __restrict__ wf,
                        const float* __restrict__ bf, float* __restrict__ out) {
    int n = blockIdx.x * 256 + threadIdx.x;
    int idx = blockIdx.y;                 // b*64 + o
    if (n >= NTOT) return;
    int b = idx >> 6, o = idx & 63;
    const float* yb = y + (size_t)b * 256 * NTOT + n;
    float acc = bf[o];
    #pragma unroll 8
    for (int c = 0; c < 256; ++c) acc += yb[(size_t)c * NTOT] * wf[o * 256 + c];
    out[(size_t)idx * NTOT + n] = acc;
}

extern "C" void kernel_launch(void* const* d_in, const int* in_sizes, int n_in,
                              void* d_out, int out_size, void* d_ws, size_t ws_size,
                              hipStream_t stream) {
    const float* x   = (const float*)d_in[0];
    const float* wk  = (const float*)d_in[1];
    const float* bk  = (const float*)d_in[2];
    const float* wq  = (const float*)d_in[3];
    const float* bq  = (const float*)d_in[4];
    const float* wv  = (const float*)d_in[5];
    const float* bv  = (const float*)d_in[6];
    const float* w1  = (const float*)d_in[7];
    const float* b1  = (const float*)d_in[8];
    const float* g1  = (const float*)d_in[9];
    const float* be1 = (const float*)d_in[10];
    const float* rm1 = (const float*)d_in[11];
    const float* rv1 = (const float*)d_in[12];
    const float* w2  = (const float*)d_in[13];
    const float* b2  = (const float*)d_in[14];
    const float* g2  = (const float*)d_in[15];
    const float* be2 = (const float*)d_in[16];
    const float* rm2 = (const float*)d_in[17];
    const float* rv2 = (const float*)d_in[18];
    const float* wf  = (const float*)d_in[19];
    const float* bf  = (const float*)d_in[20];
    float* out = (float*)d_out;

    size_t off = 0;
    auto take = [&](size_t bytes) -> char* {
        char* p = (char*)d_ws + off;
        off += (bytes + 255) & ~(size_t)255;
        return p;
    };
    const size_t N = NTOT;
    unsigned short* kb  = (unsigned short*)take(8 * N * 32 * 2);   // bf16 K   [bh][n][32]
    unsigned short* qb  = (unsigned short*)take(8 * N * 32 * 2);   // bf16 Q^T [bh][m][32]
    float*  vf   = (float*)take(2 * 128 * N * 4);                  // f32 V    [b][128][N]
    unsigned short* vtb = (unsigned short*)take(2 * 128 * N * 2);  // bf16 V/colsum
    float*  cmax = (float*)take(8 * N * 4);
    float*  csum = (float*)take(8 * N * 4);
    float*  av_t = (float*)take(2 * 128 * N * 4);                  // f32 av^T [bh][32][N]
    unsigned short* awt = (unsigned short*)take(2 * 256 * N * 2);  // bf16 avw/colsum [b][256][N]
    float*  av2t = (float*)take(2 * 256 * N * 4);                  // f32 av2^T [bh][64][N]
    float*  y    = (float*)take(2 * 256 * N * 4);                  // f32 y [b][256][N]

    dim3 blk256(256), blk128(128);
    k_qkv     <<<dim3(13, 256), blk256, 0, stream>>>(x, wk, bk, wq, bq, wv, bv, kb, qb, vf);
    k_colstats<<<dim3(49, 8),   blk128, 0, stream>>>(kb, qb, cmax, csum);
    k_vscale  <<<dim3(13, 256), blk256, 0, stream>>>(vf, csum, vtb);
    k_av      <<<dim3(49, 8),   blk128, 0, stream>>>(kb, qb, vtb, cmax, av_t);
    k_avw     <<<dim3(13, 512), blk256, 0, stream>>>(av_t, w1, b1, g1, be1, rm1, rv1, csum, awt);
    k_av2     <<<dim3(49, 8),   blk128, 0, stream>>>(kb, qb, awt, cmax, av2t);
    k_avw2    <<<dim3(13, 512), blk256, 0, stream>>>(av2t, w2, b2, g2, be2, rm2, rv2, y);
    k_final   <<<dim3(13, 128), blk256, 0, stream>>>(y, wf, bf, out);
}
